// TransformerBlock_9758165697319
// MI455X (gfx1250) — matrix operand order repaired and checked by reading, not compiled, not run on hardware
//
#include <hip/hip_runtime.h>
#include <hip/hip_bf16.h>

// ---------------------------------------------------------------------------
// Types for CDNA5 WMMA (wave32): A/B = 16 x bf16 (8 VGPRs), C/D = 8 x f32.
// ---------------------------------------------------------------------------
typedef __attribute__((ext_vector_type(16))) __bf16 v16bf;
typedef __attribute__((ext_vector_type(8)))  float  v8f;
typedef int v4i __attribute__((vector_size(16)));   // matches async-LDS builtin

struct __align__(16) U4 { unsigned x, y, z, w; };

union FragAB {
  v16bf v;
  unsigned short s[16];
  U4 q[2];
};

union U4S {
  U4 q;
  unsigned short s[8];
};

__device__ __forceinline__ unsigned short f2bf(float f) {
  unsigned u = __float_as_uint(f);
  u += 0x7FFFu + ((u >> 16) & 1u);   // round-to-nearest-even
  return (unsigned short)(u >> 16);
}

__device__ __forceinline__ v8f wmma_bf16(v16bf a, v16bf b, v8f c) {
  // D = A(16x32 bf16) x B(32x16 bf16) + C(16x16 f32)
  return __builtin_amdgcn_wmma_f32_16x16x32_bf16(false, a, false, b, (short)0, c,
                                                 false, false);
}

// gfx1250 async global->LDS path, guarded so the file compiles on any toolchain
#if defined(__has_builtin)
#if __has_builtin(__builtin_amdgcn_global_load_async_to_lds_b128)
#define HAVE_ASYNC_LDS 1
#endif
#endif
#ifndef HAVE_ASYNC_LDS
#define HAVE_ASYNC_LDS 0
#endif

#define AS1 __attribute__((address_space(1)))
#define AS3 __attribute__((address_space(3)))

#if HAVE_ASYNC_LDS
__device__ __forceinline__ void async_b128(const void* g, void* l) {
  // Param types: v4i* in AS1 (global src), v4i* in AS3 (LDS dst), imm offset,
  // cpol. (HIP prints AS1 as "__device__" in diagnostics.)
  __builtin_amdgcn_global_load_async_to_lds_b128((AS1 v4i*)g, (AS3 v4i*)l, 0, 0);
}
__device__ __forceinline__ void wait_async0() {
#if __has_builtin(__builtin_amdgcn_s_wait_asynccnt)
  __builtin_amdgcn_s_wait_asynccnt(0);
#else
  asm volatile("s_wait_asynccnt 0" ::: "memory");
#endif
}
#endif

// Problem constants
#define BATCH  128
#define TSEQ   256
#define BT     32768      // BATCH * TSEQ
#define CDIM   384
#define NHEAD  6
#define HS     64
#define QKVN   1152       // 3 * CDIM
#define FFN    1536       // 4 * CDIM

// ---------------------------------------------------------------------------
// Weight repack / convert kernels (run once per launch; tiny)
// ---------------------------------------------------------------------------
__global__ void to_bf16_kernel(const float* __restrict__ src,
                               unsigned short* __restrict__ dst, int n) {
  int i = blockIdx.x * 256 + threadIdx.x;
  if (i < n) dst[i] = f2bf(src[i]);
}

// Wq/Wk/Wv: [H, C, hs] f32 -> Wqkv bf16 [C, 1152], col = sec*384 + h*64 + d
__global__ void repack_qkv_kernel(const float* __restrict__ Wq,
                                  const float* __restrict__ Wk,
                                  const float* __restrict__ Wv,
                                  unsigned short* __restrict__ Wqkv) {
  int idx = blockIdx.x * 256 + threadIdx.x;
  if (idx >= NHEAD * CDIM * HS) return;
  int h = idx / (CDIM * HS);
  int rem = idx % (CDIM * HS);
  int c = rem / HS;
  int d = rem % HS;
  int col = h * HS + d;
  size_t base = (size_t)c * QKVN + col;
  Wqkv[base]           = f2bf(Wq[idx]);
  Wqkv[base + CDIM]    = f2bf(Wk[idx]);
  Wqkv[base + 2*CDIM]  = f2bf(Wv[idx]);
}

__global__ void pack_bias_kernel(const float* __restrict__ bq,
                                 const float* __restrict__ bk,
                                 const float* __restrict__ bv,
                                 float* __restrict__ bqkv) {
  int i = blockIdx.x * 256 + threadIdx.x;
  if (i < CDIM) {
    bqkv[i]           = bq[i];   // bq flat [H,hs] == col order h*64+d
    bqkv[i + CDIM]    = bk[i];
    bqkv[i + 2*CDIM]  = bv[i];
  }
}

// ---------------------------------------------------------------------------
// LayerNorm -> bf16. One block (128 threads = 4 waves) per row of 384.
// ---------------------------------------------------------------------------
__global__ __launch_bounds__(128) void ln_bf16_kernel(
    const float* __restrict__ x, const float* __restrict__ g,
    const float* __restrict__ be, unsigned short* __restrict__ out) {
  int row = blockIdx.x;
  int tid = threadIdx.x;
  const float* xr = x + (size_t)row * CDIM;
  float a0 = xr[tid], a1 = xr[tid + 128], a2 = xr[tid + 256];
  float s  = a0 + a1 + a2;
  float s2 = a0 * a0 + a1 * a1 + a2 * a2;
  #pragma unroll
  for (int off = 16; off; off >>= 1) {
    s  += __shfl_xor(s, off, 32);
    s2 += __shfl_xor(s2, off, 32);
  }
  __shared__ float red[2][4];
  int w = tid >> 5, lane = tid & 31;
  if (lane == 0) { red[0][w] = s; red[1][w] = s2; }
  __syncthreads();
  s  = red[0][0] + red[0][1] + red[0][2] + red[0][3];
  s2 = red[1][0] + red[1][1] + red[1][2] + red[1][3];
  float mu  = s * (1.0f / CDIM);
  float var = s2 * (1.0f / CDIM) - mu * mu;
  float rs  = rsqrtf(var + 1e-5f);
  unsigned short* orow = out + (size_t)row * CDIM;
  orow[tid]       = f2bf((a0 - mu) * rs * g[tid]       + be[tid]);
  orow[tid + 128] = f2bf((a1 - mu) * rs * g[tid + 128] + be[tid + 128]);
  orow[tid + 256] = f2bf((a2 - mu) * rs * g[tid + 256] + be[tid + 256]);
}

// ---------------------------------------------------------------------------
// Tiled bf16 WMMA GEMM: out[M,N] = A[M,K](bf16) * B[K,N](bf16) + bias
// Block tile 128x128, K-chunk 32, double-buffered LDS (one barrier/chunk).
// 8 waves; wave (wm,wn) owns 32x64 = 2x4 tiles -> 8 v_wmma per chunk.
// Staging: b128 per thread; A tile via async global->LDS (gfx1250) when
// available, B tile transposed to [n][k] in LDS so B fragments are b128 loads.
// Epilogue: optional exact GELU, optional f32 residual, f32 and/or bf16 store.
// All dims are exact multiples of the tiles for this problem.
// ---------------------------------------------------------------------------
__global__ __launch_bounds__(256) void gemm_bf16_kernel(
    const unsigned short* __restrict__ A, const unsigned short* __restrict__ B,
    const float* __restrict__ bias, const float* __restrict__ res,
    float* __restrict__ outF, unsigned short* __restrict__ outB,
    int M, int N, int K, int act_gelu) {
  __shared__ __align__(16) unsigned short As[2][128][48];  // [m][k], 96B rows
  __shared__ __align__(16) unsigned short Bs[2][128][48];  // transposed: [n][k]

  const int tid  = threadIdx.x;
  const int lane = tid & 31, w = tid >> 5;
  const int lm = lane & 15, g = lane >> 4;
  const int wm = w >> 1, wn = w & 1;
  const int bx = blockIdx.x, by = blockIdx.y;

  v8f vzero = {};
  v8f acc[2][4];
  #pragma unroll
  for (int i = 0; i < 2; ++i)
    #pragma unroll
    for (int j = 0; j < 4; ++j) acc[i][j] = vzero;

  // A staging: thread covers rows ar0 and ar0+64, 16B chunk ac8 within chunk
  const int ar0 = tid >> 2;            // 0..63
  const int ac8 = (tid & 3) * 8;       // element offset 0/8/16/24
  const unsigned short* agp = A + (size_t)(by * 128 + ar0) * K + ac8;
  const size_t a2off = (size_t)64 * K; // second A chunk row offset

  // B staging: k rows br0 and br0+16, 8 consecutive cols at bc8
  const int br0 = tid >> 4;            // 0..15
  const int bc8 = (tid & 15) * 8;      // col offset
  const unsigned short* bgp = B + (size_t)br0 * N + bx * 128 + bc8;
  const size_t b2off = (size_t)16 * N; // second B chunk row offset

  const int nchunks = K >> 5;

  auto stageA = [&](int buf, int kc) {
    const unsigned short* p = agp + (size_t)kc * 32;
#if HAVE_ASYNC_LDS
    async_b128(p,         &As[buf][ar0][ac8]);
    async_b128(p + a2off, &As[buf][ar0 + 64][ac8]);
#else
    *(U4*)&As[buf][ar0][ac8]      = *(const U4*)p;
    *(U4*)&As[buf][ar0 + 64][ac8] = *(const U4*)(p + a2off);
#endif
  };
  auto stageB = [&](int buf, int kc) {
    const unsigned short* p = bgp + (size_t)kc * 32 * N;
    U4S u0, u1;
    u0.q = *(const U4*)p;
    u1.q = *(const U4*)(p + b2off);
    #pragma unroll
    for (int j = 0; j < 8; ++j) Bs[buf][bc8 + j][br0]      = u0.s[j];
    #pragma unroll
    for (int j = 0; j < 8; ++j) Bs[buf][bc8 + j][br0 + 16] = u1.s[j];
  };

  stageA(0, 0);
  stageB(0, 0);
#if HAVE_ASYNC_LDS
  wait_async0();
#endif
  __syncthreads();

  for (int kc = 0; kc < nchunks; ++kc) {
    const int cur = kc & 1;
    const bool more = (kc + 1) < nchunks;
    if (more) {
      stageA(cur ^ 1, kc + 1);
      stageB(cur ^ 1, kc + 1);
    }

    // A fragment: lane m=lm, element e -> k = 8g + e (+8 if e>=8)
    FragAB af[2];
    #pragma unroll
    for (int ms = 0; ms < 2; ++ms) {
      const unsigned short* p = &As[cur][wm * 32 + ms * 16 + lm][0];
      af[ms].q[0] = *(const U4*)(p + 8 * g);
      af[ms].q[1] = *(const U4*)(p + 16 + 8 * g);
    }
    // B fragment: lane n=lm, element e -> k = 8g + e (+8 if e>=8)
    FragAB bf[4];
    #pragma unroll
    for (int ns = 0; ns < 4; ++ns) {
      const unsigned short* p = &Bs[cur][wn * 64 + ns * 16 + lm][0];
      bf[ns].q[0] = *(const U4*)(p + 8 * g);
      bf[ns].q[1] = *(const U4*)(p + 16 + 8 * g);
    }
    #pragma unroll
    for (int ms = 0; ms < 2; ++ms)
      #pragma unroll
      for (int ns = 0; ns < 4; ++ns)
        acc[ms][ns] = wmma_bf16(af[ms].v, bf[ns].v, acc[ms][ns]);

    if (more) {
#if HAVE_ASYNC_LDS
      wait_async0();
#endif
      __syncthreads();
    }
  }

  // Epilogue. C/D layout: lane col = lm, element e -> row = e + 8g
  #pragma unroll
  for (int ms = 0; ms < 2; ++ms) {
    int row0 = by * 128 + wm * 32 + ms * 16 + 8 * g;
    #pragma unroll
    for (int ns = 0; ns < 4; ++ns) {
      int col = bx * 128 + wn * 64 + ns * 16 + lm;
      float bcol = bias[col];
      #pragma unroll
      for (int e = 0; e < 8; ++e) {
        float val = acc[ms][ns][e] + bcol;
        if (act_gelu) val = 0.5f * val * (1.0f + erff(val * 0.70710678118f));
        size_t idx = (size_t)(row0 + e) * N + col;
        if (res)  val += res[idx];
        if (outF) outF[idx] = val;
        if (outB) outB[idx] = f2bf(val);
      }
    }
  }
}

// ---------------------------------------------------------------------------
// Attention: one block = one (b, h, 32-row Q strip). 8 waves.
// Phase 1: S[32,256] = (Q K^T) * 1/8 via WMMA into LDS (f32).
// Phase 2: per-row softmax (wave32 shuffle reductions), P stored bf16 in
//          place over S.
// Phase 3: O[32,64] = P V via WMMA; written bf16 head-concat to att[M,384].
// QKV layout: row (b*T + t) of 1152: [Q(h*64+d) | K +384 | V +768].
// ---------------------------------------------------------------------------
#define SROW 260                    // f32 stride of an S row (65*16B: aligned)
#define ATT_SMEM_BYTES (32 * SROW * 4)

__global__ __launch_bounds__(256) void attn_kernel(
    const unsigned short* __restrict__ qkv, unsigned short* __restrict__ attout) {
  extern __shared__ float Ss[];     // [32][SROW]
  const int bh = blockIdx.y;
  const int b  = bh / NHEAD, h = bh % NHEAD;
  const int t0 = blockIdx.x * 32;
  const int tid = threadIdx.x;
  const int lane = tid & 31, w = tid >> 5;
  const int lm = lane & 15, g = lane >> 4;
  const size_t rowbase = (size_t)b * TSEQ;   // first QKV row of this batch

  // ---- Phase 1: S = Q K^T * scale ----
  {
    const int mt = w & 1;           // which 16-row Q tile
    const int nq = w >> 1;          // which 64-key quadrant
    const unsigned short* qrow =
        qkv + (rowbase + t0 + mt * 16 + lm) * QKVN + h * HS;
    FragAB qf[2];
    #pragma unroll
    for (int ks = 0; ks < 2; ++ks) {
      qf[ks].q[0] = *(const U4*)(qrow + 32 * ks + 8 * g);
      qf[ks].q[1] = *(const U4*)(qrow + 32 * ks + 16 + 8 * g);
    }
    #pragma unroll
    for (int nt = 0; nt < 4; ++nt) {
      int key0 = nq * 64 + nt * 16;
      const unsigned short* krow =
          qkv + (rowbase + key0 + lm) * QKVN + CDIM + h * HS;
      v8f c = {};
      #pragma unroll
      for (int ks = 0; ks < 2; ++ks) {
        FragAB kf;  // B frag: lane n = key lm, element e -> d = 32ks + 8g + e (+8 if e>=8)
        kf.q[0] = *(const U4*)(krow + 32 * ks + 8 * g);
        kf.q[1] = *(const U4*)(krow + 32 * ks + 16 + 8 * g);
        c = wmma_bf16(qf[ks].v, kf.v, c);
      }
      #pragma unroll
      for (int e = 0; e < 8; ++e)
        Ss[(mt * 16 + e + 8 * g) * SROW + key0 + lm] = c[e] * 0.125f;
    }
  }
  __syncthreads();

  // ---- Phase 2: softmax over 256 keys; P (bf16) written in place ----
  #pragma unroll
  for (int r4 = 0; r4 < 4; ++r4) {
    int r = w * 4 + r4;
    float* srow = Ss + (size_t)r * SROW;
    float vals[8];
    #pragma unroll
    for (int j = 0; j < 8; ++j) vals[j] = srow[lane + 32 * j];
    float mx = vals[0];
    #pragma unroll
    for (int j = 1; j < 8; ++j) mx = fmaxf(mx, vals[j]);
    #pragma unroll
    for (int off = 16; off; off >>= 1) mx = fmaxf(mx, __shfl_xor(mx, off, 32));
    float sum = 0.0f;
    #pragma unroll
    for (int j = 0; j < 8; ++j) { vals[j] = __expf(vals[j] - mx); sum += vals[j]; }
    #pragma unroll
    for (int off = 16; off; off >>= 1) sum += __shfl_xor(sum, off, 32);
    float inv = 1.0f / sum;
    unsigned short* prow = (unsigned short*)srow;
    #pragma unroll
    for (int j = 0; j < 8; ++j) prow[lane + 32 * j] = f2bf(vals[j] * inv);
  }
  __syncthreads();

  // ---- Phase 3: O = P V ----
  {
    const int m2 = w >> 2;          // 16-row tile
    const int nt = w & 3;           // 16-col (head dim) tile
    v8f o = {};
    #pragma unroll
    for (int ks = 0; ks < 8; ++ks) {  // K = 256 keys, 32 per step
      FragAB pf;  // A frag from LDS (bf16 P): row = m2*16+lm, k = 32ks+16g+e
      const unsigned short* pr =
          (const unsigned short*)((const char*)Ss +
                                  (size_t)(m2 * 16 + lm) * (SROW * 4));
      pf.q[0] = *(const U4*)(pr + 32 * ks + 16 * g);
      pf.q[1] = *(const U4*)(pr + 32 * ks + 16 * g + 8);
      FragAB vf;  // B frag: lane n = d (lm), element e -> key s = 32ks+16g+e
      const unsigned short* vptr =
          qkv + (rowbase + 32 * ks + 16 * g) * QKVN + 2 * CDIM + h * HS +
          nt * 16 + lm;
      #pragma unroll
      for (int e = 0; e < 16; ++e) vf.s[e] = vptr[(size_t)e * QKVN];
      o = wmma_bf16(pf.v, vf.v, o);
    }
    #pragma unroll
    for (int e = 0; e < 8; ++e) {
      size_t r = rowbase + t0 + m2 * 16 + e + 8 * g;
      attout[r * CDIM + h * HS + nt * 16 + lm] = f2bf(o[e]);
    }
  }
}

// ---------------------------------------------------------------------------
// Host-side orchestration
// ---------------------------------------------------------------------------
extern "C" void kernel_launch(void* const* d_in, const int* in_sizes, int n_in,
                              void* d_out, int out_size, void* d_ws,
                              size_t ws_size, hipStream_t stream) {
  (void)in_sizes; (void)n_in; (void)out_size; (void)ws_size;
  const float* x   = (const float*)d_in[0];
  const float* Wq  = (const float*)d_in[1];
  const float* bq  = (const float*)d_in[2];
  const float* Wk  = (const float*)d_in[3];
  const float* bk  = (const float*)d_in[4];
  const float* Wv  = (const float*)d_in[5];
  const float* bv  = (const float*)d_in[6];
  const float* Wo  = (const float*)d_in[7];
  const float* bo  = (const float*)d_in[8];
  const float* W1  = (const float*)d_in[9];
  const float* b1  = (const float*)d_in[10];
  const float* W2  = (const float*)d_in[11];
  const float* b2  = (const float*)d_in[12];
  const float* g1  = (const float*)d_in[13];
  const float* be1 = (const float*)d_in[14];
  const float* g2  = (const float*)d_in[15];
  const float* be2 = (const float*)d_in[16];

  // Workspace carve (256B aligned bumps)
  char* ws = (char*)d_ws;
  size_t off = 0;
  auto carve = [&](size_t bytes) -> char* {
    off = (off + 255) & ~(size_t)255;
    char* p = ws + off;
    off += bytes;
    return p;
  };
  unsigned short* h_bf    = (unsigned short*)carve((size_t)BT * CDIM * 2);
  unsigned short* wqkv_bf = (unsigned short*)carve((size_t)CDIM * QKVN * 2);
  float*          bqkv    = (float*)carve((size_t)QKVN * 4);
  unsigned short* wo_bf   = (unsigned short*)carve((size_t)CDIM * CDIM * 2);
  unsigned short* w1_bf   = (unsigned short*)carve((size_t)CDIM * FFN * 2);
  unsigned short* w2_bf   = (unsigned short*)carve((size_t)FFN * CDIM * 2);
  float*          x1      = (float*)carve((size_t)BT * CDIM * 4);
  // Shared region: QKV [M,1152] + att [M,384] == later reused as F1 [M,1536]
  unsigned short* region  = (unsigned short*)carve((size_t)BT * FFN * 2);
  unsigned short* qkv_bf  = region;
  unsigned short* att_bf  = region + (size_t)BT * QKVN;
  unsigned short* f1_bf   = region;

  // --- weight prep ---
  to_bf16_kernel<<<(CDIM * CDIM + 255) / 256, 256, 0, stream>>>(Wo, wo_bf, CDIM * CDIM);
  to_bf16_kernel<<<(CDIM * FFN + 255) / 256, 256, 0, stream>>>(W1, w1_bf, CDIM * FFN);
  to_bf16_kernel<<<(FFN * CDIM + 255) / 256, 256, 0, stream>>>(W2, w2_bf, FFN * CDIM);
  repack_qkv_kernel<<<(NHEAD * CDIM * HS + 255) / 256, 256, 0, stream>>>(Wq, Wk, Wv, wqkv_bf);
  pack_bias_kernel<<<2, 256, 0, stream>>>(bq, bk, bv, bqkv);

  // --- ln1 -> h ---
  ln_bf16_kernel<<<BT, 128, 0, stream>>>(x, g1, be1, h_bf);
  // --- QKV = h * Wqkv + bqkv  (store bf16) ---
  gemm_bf16_kernel<<<dim3(QKVN / 128, BT / 128), 256, 0, stream>>>(
      h_bf, wqkv_bf, bqkv, nullptr, nullptr, qkv_bf, BT, QKVN, CDIM, 0);
  // --- attention -> att (bf16, head-concat) ---
  attn_kernel<<<dim3(TSEQ / 32, BATCH * NHEAD), 256, ATT_SMEM_BYTES, stream>>>(
      qkv_bf, att_bf);
  // --- x1 = x + att * Wo + bo (f32) ---
  gemm_bf16_kernel<<<dim3(CDIM / 128, BT / 128), 256, 0, stream>>>(
      att_bf, wo_bf, bo, x, x1, nullptr, BT, CDIM, CDIM, 0);
  // --- ln2 -> h2 (reuse h_bf) ---
  ln_bf16_kernel<<<BT, 128, 0, stream>>>(x1, g2, be2, h_bf);
  // --- f1 = gelu(h2 * W1 + b1) (bf16; overwrites qkv/att region) ---
  gemm_bf16_kernel<<<dim3(FFN / 128, BT / 128), 256, 0, stream>>>(
      h_bf, w1_bf, b1, nullptr, nullptr, f1_bf, BT, FFN, CDIM, 1);
  // --- out = x1 + f1 * W2 + b2 (f32) ---
  gemm_bf16_kernel<<<dim3(CDIM / 128, BT / 128), 256, 0, stream>>>(
      f1_bf, w2_bf, b2, x1, (float*)d_out, nullptr, BT, CDIM, FFN, 0);
}
